// AttentionHead_59631325937668
// MI455X (gfx1250) — matrix operand order repaired and checked by reading, not compiled, not run on hardware
//
#include <hip/hip_runtime.h>
#include <stdint.h>

#define B_  8
#define S_  2048
#define DIN 768
#define DK  64

typedef __attribute__((ext_vector_type(16))) __bf16 v16bf;
typedef __attribute__((ext_vector_type(8)))  float  v8f;
typedef __attribute__((ext_vector_type(2)))  __bf16 bf16x2;

union ABReg { v16bf v; uint4 q[2]; };

// native f32 -> bf16 (v_cvt_pk_bf16_f32 on gfx1250)
static __device__ __forceinline__ unsigned short bfbits(float f) {
  return __builtin_bit_cast(unsigned short, (__bf16)f);
}
static __device__ __forceinline__ unsigned pk2(float lo, float hi) {
  bf16x2 p = {(__bf16)lo, (__bf16)hi};
  return __builtin_bit_cast(unsigned, p);
}
static __device__ __forceinline__ v8f wmma_bf16(v16bf a, v16bf b, v8f c) {
  return __builtin_amdgcn_wmma_f32_16x16x32_bf16(
      /*neg_a=*/false, a, /*neg_b=*/false, b,
      /*c_mod=*/(short)0, c, /*reuse_a=*/false, /*reuse_b=*/false);
}
static __device__ __forceinline__ void lds_fence() {
  asm volatile("s_wait_dscnt 0x0" ::: "memory");
}
static __device__ __forceinline__ void async_fence() {
  asm volatile("s_wait_asynccnt 0x0" ::: "memory");
}
// gfx1250 async global -> LDS copy (ASYNCcnt-tracked, no VGPR round trip).
// VDST = per-lane LDS byte offset, VADDR = 64-bit global address, SADDR = off.
static __device__ __forceinline__ void async_load_b128(unsigned lds_off,
                                                       const void* gptr) {
  asm volatile("global_load_async_to_lds_b128 %0, %1, off"
               :: "v"(lds_off), "v"(gptr) : "memory");
}
static __device__ __forceinline__ unsigned lds_offset(const void* p) {
  return (unsigned)(uintptr_t)p;  // low dword of flat LDS pointer = AS3 offset
}

// ---------------------------------------------------------------------------
// Pass 1: proj[r, o] = (sum_i x[r,i] * W[o,i] + bias[o]) * scale   -> bf16
// ---------------------------------------------------------------------------
__global__ __launch_bounds__(128)
void proj_kernel(const float* __restrict__ x, const float* __restrict__ W,
                 const float* __restrict__ bias, unsigned short* __restrict__ out,
                 float scale) {
  __shared__ __align__(16) unsigned short Wlds[DK * 32];  // [o][i] 64x32 bf16

  const int t    = threadIdx.x;
  const int lane = t & 31;
  const int wave = t >> 5;
  const int n    = lane & 15;
  const int h    = lane >> 4;
  const int myrow = blockIdx.x * 64 + wave * 16;

  v8f acc[4] = {v8f{}, v8f{}, v8f{}, v8f{}};

  for (int it = 0; it < DIN; it += 32) {
    __syncthreads();
    {  // cooperative W tile load + f32->bf16 convert
      const int o  = t >> 1;
      const int ic = (t & 1) * 16;
      const float4* wp = (const float4*)(W + (size_t)o * DIN + it + ic);
      float4 w0 = wp[0], w1 = wp[1], w2 = wp[2], w3 = wp[3];
      uint4 lo = make_uint4(pk2(w0.x, w0.y), pk2(w0.z, w0.w),
                            pk2(w1.x, w1.y), pk2(w1.z, w1.w));
      uint4 hi = make_uint4(pk2(w2.x, w2.y), pk2(w2.z, w2.w),
                            pk2(w3.x, w3.y), pk2(w3.z, w3.w));
      *(uint4*)(Wlds + o * 32 + ic)     = lo;
      *(uint4*)(Wlds + o * 32 + ic + 8) = hi;
    }
    __syncthreads();

    ABReg a;
    {
      const float* xr = x + (size_t)(myrow + n) * DIN + it;
      const float4* p0 = (const float4*)(xr + 8 * h);
      float4 x0 = p0[0], x1 = p0[1];
      const float4* p1 = (const float4*)(xr + 16 + 8 * h);
      float4 x2 = p1[0], x3 = p1[1];
      a.q[0] = make_uint4(pk2(x0.x, x0.y), pk2(x0.z, x0.w),
                          pk2(x1.x, x1.y), pk2(x1.z, x1.w));
      a.q[1] = make_uint4(pk2(x2.x, x2.y), pk2(x2.z, x2.w),
                          pk2(x3.x, x3.y), pk2(x3.z, x3.w));
    }
    ABReg b0, b1, b2, b3;
    {
      const unsigned short* wl = Wlds + n * 32 + 8 * h;
      b0.q[0] = *(const uint4*)(wl);        b0.q[1] = *(const uint4*)(wl + 16);
      b1.q[0] = *(const uint4*)(wl + 512);  b1.q[1] = *(const uint4*)(wl + 528);
      b2.q[0] = *(const uint4*)(wl + 1024); b2.q[1] = *(const uint4*)(wl + 1040);
      b3.q[0] = *(const uint4*)(wl + 1536); b3.q[1] = *(const uint4*)(wl + 1552);
    }
    acc[0] = wmma_bf16(a.v, b0.v, acc[0]);
    acc[1] = wmma_bf16(a.v, b1.v, acc[1]);
    acc[2] = wmma_bf16(a.v, b2.v, acc[2]);
    acc[3] = wmma_bf16(a.v, b3.v, acc[3]);
  }

#pragma unroll
  for (int j = 0; j < 4; ++j) {
    float bj = bias[j * 16 + n];
#pragma unroll
    for (int r = 0; r < 8; ++r) {
      int m = r + 8 * h;
      out[(size_t)(myrow + m) * DK + j * 16 + n] = bfbits((acc[j][r] + bj) * scale);
    }
  }
}

// ---------------------------------------------------------------------------
// Pass 2: flash attention, double-buffered pipeline.
// K tiles: async global->LDS (ASYNCcnt). V tiles: regs + packed transpose.
// ---------------------------------------------------------------------------
__global__ __launch_bounds__(128)
void attn_kernel(const unsigned short* __restrict__ qb,
                 const unsigned short* __restrict__ kb,
                 const unsigned short* __restrict__ vb,
                 const uint8_t* __restrict__ mask,
                 float* __restrict__ outp) {
  __shared__ __align__(16) unsigned short Kt[2][32 * DK];   // [key][dim]
  __shared__ __align__(16) unsigned short Vt[2][DK * 32];   // [dim][key]
  __shared__ __align__(16) unsigned short Pl[4 * 16 * 32];  // per-wave P

  const int t    = threadIdx.x;
  const int lane = t & 31;
  const int wave = t >> 5;
  const int n    = lane & 15;
  const int h    = lane >> 4;

  const int tiles_per_b = S_ / 64;
  const int b     = blockIdx.x / tiles_per_b;
  const int qbase = (blockIdx.x % tiles_per_b) * 64;
  const int qrow  = b * S_ + qbase + wave * 16;

  // Q A-matrices, resident all kernel
  ABReg aq0, aq1;
  {
    const unsigned short* qr = qb + (size_t)(qrow + n) * DK;
    aq0.q[0] = *(const uint4*)(qr + 8 * h);
    aq0.q[1] = *(const uint4*)(qr + 16 + 8 * h);
    aq1.q[0] = *(const uint4*)(qr + 32 + 8 * h);
    aq1.q[1] = *(const uint4*)(qr + 48 + 8 * h);
  }
  const uint8_t* mrow = mask + (size_t)b * S_;

  v8f acc[4] = {v8f{}, v8f{}, v8f{}, v8f{}};
  float rmax[8], rsum[8];
#pragma unroll
  for (int r = 0; r < 8; ++r) { rmax[r] = -__builtin_inff(); rsum[r] = 0.f; }

  unsigned short* pw = Pl + wave * 16 * 32;

  // K async-copy role: thread -> (key, 16-dim chunk)
  const int kkey = t >> 2;
  const int kdc  = (t & 3) * 16;
  const size_t ksrc0 = (size_t)(b * S_ + kkey) * DK + kdc;
  // V transpose role: thread -> (key pair, 8-dim chunk)
  const int vkey = (t & 15) * 2;
  const int vdc  = (t >> 4) * 8;
  const size_t vsrc0 = (size_t)(b * S_ + vkey) * DK + vdc;

  uint4 rv0, rv1;
  auto commitV = [&](int buf) {
    const unsigned short* e0 = (const unsigned short*)&rv0;
    const unsigned short* e1 = (const unsigned short*)&rv1;
#pragma unroll
    for (int e = 0; e < 8; ++e) {
      unsigned d = (unsigned)e0[e] | ((unsigned)e1[e] << 16);  // keys (vkey,vkey+1)
      *(unsigned*)(&Vt[buf][(vdc + e) * 32 + vkey]) = d;
    }
  };

  // prologue: tile 0
  async_load_b128(lds_offset(&Kt[0][kkey * DK + kdc]), kb + ksrc0);
  rv0 = *(const uint4*)(vb + vsrc0);
  rv1 = *(const uint4*)(vb + vsrc0 + DK);
  commitV(0);
  async_fence();
  __syncthreads();

  int cur = 0;
  for (int kb0 = 0; kb0 < S_; kb0 += 32) {
    const bool more = (kb0 + 32) < S_;
    if (more) {  // next tile: async K straight into LDS, V into regs
      const int nxt = cur ^ 1;
      async_load_b128(lds_offset(&Kt[nxt][kkey * DK + kdc]),
                      kb + ksrc0 + (size_t)(kb0 + 32) * DK);
      rv0 = *(const uint4*)(vb + vsrc0 + (size_t)(kb0 + 32) * DK);
      rv1 = *(const uint4*)(vb + vsrc0 + (size_t)(kb0 + 32) * DK + DK);
      if (kb0 + 64 < S_) {  // global_prefetch_b8 two tiles ahead
        __builtin_prefetch(kb + ksrc0 + (size_t)(kb0 + 64) * DK, 0, 1);
        __builtin_prefetch(vb + vsrc0 + (size_t)(kb0 + 64) * DK, 0, 1);
      }
    }

    // ---- scores: preload 4 B fragments, then 4 chained WMMAs ----
    const unsigned short* kt = Kt[cur];
    ABReg bk00, bk01, bk10, bk11;
    {
      const unsigned short* kr0 = kt + n * DK;
      const unsigned short* kr1 = kt + (16 + n) * DK;
      bk00.q[0] = *(const uint4*)(kr0 + 8 * h);      bk00.q[1] = *(const uint4*)(kr0 + 8 * h + 16);
      bk01.q[0] = *(const uint4*)(kr0 + 32 + 8 * h); bk01.q[1] = *(const uint4*)(kr0 + 48 + 8 * h);
      bk10.q[0] = *(const uint4*)(kr1 + 8 * h);      bk10.q[1] = *(const uint4*)(kr1 + 8 * h + 16);
      bk11.q[0] = *(const uint4*)(kr1 + 32 + 8 * h); bk11.q[1] = *(const uint4*)(kr1 + 48 + 8 * h);
    }
    v8f c0 = {}, c1 = {};
    c0 = wmma_bf16(aq0.v, bk00.v, c0);
    c1 = wmma_bf16(aq0.v, bk10.v, c1);
    c0 = wmma_bf16(aq1.v, bk01.v, c0);
    c1 = wmma_bf16(aq1.v, bk11.v, c1);

    // ---- mask (True -> -1e18) + online softmax ----
    const bool m0 = mrow[kb0 + n] != 0;
    const bool m1 = mrow[kb0 + 16 + n] != 0;
#pragma unroll
    for (int r = 0; r < 8; ++r) {
      float s0 = m0 ? -1e18f : c0[r];
      float s1 = m1 ? -1e18f : c1[r];
      float tmax = fmaxf(s0, s1);
#pragma unroll
      for (int off = 1; off < 16; off <<= 1) tmax = fmaxf(tmax, __shfl_xor(tmax, off, 32));
      float mn   = fmaxf(rmax[r], tmax);
      float corr = __expf(rmax[r] - mn);
      rmax[r] = mn;
      float p0 = __expf(s0 - mn);
      float p1 = __expf(s1 - mn);
      float ls = p0 + p1;
#pragma unroll
      for (int off = 1; off < 16; off <<= 1) ls += __shfl_xor(ls, off, 32);
      rsum[r] = rsum[r] * corr + ls;
      acc[0][r] *= corr; acc[1][r] *= corr; acc[2][r] *= corr; acc[3][r] *= corr;
      pw[(r + 8 * h) * 32 + n]      = bfbits(p0);
      pw[(r + 8 * h) * 32 + 16 + n] = bfbits(p1);
    }
    lds_fence();  // P stores visible before cross-lane A-fragment reads

    // ---- out += P(16x32) x V(32x64) ----
    const unsigned short* vt = Vt[cur];
    ABReg ap, bv0, bv1, bv2, bv3;
    ap.q[0] = *(const uint4*)(pw + n * 32 + 8 * h);
    ap.q[1] = *(const uint4*)(pw + n * 32 + 16 + 8 * h);
    {
      const unsigned short* vr = vt + n * 32 + 8 * h;
      bv0.q[0] = *(const uint4*)(vr);        bv0.q[1] = *(const uint4*)(vr + 16);
      bv1.q[0] = *(const uint4*)(vr + 512);  bv1.q[1] = *(const uint4*)(vr + 528);
      bv2.q[0] = *(const uint4*)(vr + 1024); bv2.q[1] = *(const uint4*)(vr + 1040);
      bv3.q[0] = *(const uint4*)(vr + 1536); bv3.q[1] = *(const uint4*)(vr + 1552);
    }
    acc[0] = wmma_bf16(ap.v, bv0.v, acc[0]);
    acc[1] = wmma_bf16(ap.v, bv1.v, acc[1]);
    acc[2] = wmma_bf16(ap.v, bv2.v, acc[2]);
    acc[3] = wmma_bf16(ap.v, bv3.v, acc[3]);

    // ---- commit next V tile, fence async K, flip buffers ----
    __syncthreads();
    if (more) commitV(cur ^ 1);
    async_fence();  // this wave's async K copies have landed in LDS
    __syncthreads();
    cur ^= 1;
  }

  // normalize and store fp32 output
#pragma unroll
  for (int j = 0; j < 4; ++j) {
#pragma unroll
    for (int r = 0; r < 8; ++r) {
      int m = r + 8 * h;
      outp[(size_t)(qrow + m) * DK + j * 16 + n] = acc[j][r] * (1.0f / rsum[r]);
    }
  }
}

// ---------------------------------------------------------------------------
extern "C" void kernel_launch(void* const* d_in, const int* in_sizes, int n_in,
                              void* d_out, int out_size, void* d_ws, size_t ws_size,
                              hipStream_t stream) {
  const float*   query = (const float*)d_in[0];
  const float*   keyx  = (const float*)d_in[1];
  const float*   valx  = (const float*)d_in[2];
  const uint8_t* mask  = (const uint8_t*)d_in[3];  // jnp.bool_ -> 1 byte/elem
  const float*   W_q   = (const float*)d_in[4];
  const float*   b_q   = (const float*)d_in[5];
  const float*   W_k   = (const float*)d_in[6];
  const float*   b_k   = (const float*)d_in[7];
  const float*   W_v   = (const float*)d_in[8];
  const float*   b_v   = (const float*)d_in[9];
  float* outp = (float*)d_out;

  const size_t N = (size_t)B_ * S_ * DK;
  unsigned short* qbuf = (unsigned short*)d_ws;
  unsigned short* kbuf = qbuf + N;
  unsigned short* vbuf = kbuf + N;

  dim3 grid(B_ * S_ / 64), block(128);
  proj_kernel<<<grid, block, 0, stream>>>(query, W_q, b_q, qbuf, 0.125f);  // 1/sqrt(64) folded
  proj_kernel<<<grid, block, 0, stream>>>(keyx,  W_k, b_k, kbuf, 1.0f);
  proj_kernel<<<grid, block, 0, stream>>>(valx,  W_v, b_v, vbuf, 1.0f);
  attn_kernel<<<grid, block, 0, stream>>>(qbuf, kbuf, vbuf, mask, outp);
}
